// Attention_12403865551261
// MI455X (gfx1250) — matrix operand order repaired and checked by reading, not compiled, not run on hardware
//
#include <hip/hip_runtime.h>

// Fused causal multi-head attention for MI455X (gfx1250, wave32, WMMA).
// B=1, S=4096, E=768, H=12, DH=64. All matmuls on v_wmma_f32_16x16x32_f16
// (f16 operands, fp32 accumulate); softmax stats fp32; flash-style fusion.
// Tile staging uses gfx1250 async global->LDS copies (ASYNCcnt) with
// double-buffered LDS so copies overlap WMMA; falls back to register
// staging if the async builtins are not declared by this toolchain.

typedef __attribute__((ext_vector_type(16))) _Float16 v16h;
typedef __attribute__((ext_vector_type(8)))  _Float16 v8h;
typedef __attribute__((ext_vector_type(4)))  _Float16 v4h;
typedef __attribute__((ext_vector_type(8)))  float    v8f;
typedef __attribute__((ext_vector_type(4)))  float    v4f;

#define S_LEN 4096
#define EMB   768
#define NH    12
#define DHD   64
#define E3    2304

// ---------------- async global->LDS staging (gfx1250) ----------------------
#if __has_builtin(__builtin_amdgcn_global_load_async_to_lds_b128)
#define ASYNC_LDS 1
#else
#define ASYNC_LDS 0
#endif

#if ASYNC_LDS
// Parameter types per hipcc diagnostic: pointers to int __vector(4) in the
// global (AS1) / LDS (AS3) address spaces.
typedef int v4i_vs __attribute__((vector_size(16)));
typedef __attribute__((address_space(1))) v4i_vs* gptr_b128;
typedef __attribute__((address_space(3))) v4i_vs* lptr_b128;
#if __has_builtin(__builtin_amdgcn_s_wait_asynccnt)
#define WAIT_ASYNC(n) __builtin_amdgcn_s_wait_asynccnt(n)
#else
#define WAIT_ASYNC(n) asm volatile("s_wait_asynccnt %0" ::"i"(n) : "memory")
#endif
#else
#define WAIT_ASYNC(n) ((void)0)
#endif

// Copy 16 bytes global -> LDS. Async path writes LDS directly (no VGPR trip).
static __device__ __forceinline__ void stage16(void* lds_dst, const void* gsrc) {
#if ASYNC_LDS
  __builtin_amdgcn_global_load_async_to_lds_b128(
      (gptr_b128)gsrc, (lptr_b128)lds_dst, /*offset=*/0, /*cpol=*/0);
#else
  *reinterpret_cast<v8h*>(lds_dst) = *reinterpret_cast<const v8h*>(gsrc);
#endif
}

// ---------------- WMMA helpers --------------------------------------------
static __device__ __forceinline__ v8f wmma_f16(v16h a, v16h b, v8f c) {
  return __builtin_amdgcn_wmma_f32_16x16x32_f16(
      /*neg_a=*/false, a, /*neg_b=*/false, b,
      /*c_mod=*/(short)0, c, /*reuse_a=*/false, /*reuse_b=*/false);
}

// A-operand fragment (16x32, row-major source, stride ld halves).
// ISA layout: lanes 0-15 row=lane,K={0..7,16..23}; lanes 16-31 K={8..15,24..31}.
static __device__ __forceinline__ v16h frag_a_16(const _Float16* base, int ld, int lane) {
  int row = lane & 15;
  int kb  = (lane >> 4) << 3;               // 0 or 8
  const _Float16* p = base + row * ld + kb;
  v8h lo = *reinterpret_cast<const v8h*>(p);
  v8h hi = *reinterpret_cast<const v8h*>(p + 16);
  v16h f;
#pragma unroll
  for (int i = 0; i < 8; ++i) { f[i] = lo[i]; f[i + 8] = hi[i]; }
  return f;
}

// B-operand fragment (32x16). Source is B^T row-major (n-major, stride ld):
// lanes 0-15 col=lane,K={0..7,16..23}; lanes 16-31 K={8..15,24..31}.
static __device__ __forceinline__ v16h frag_b_16(const _Float16* base, int ld, int lane) {
  int col = lane & 15;
  int kb  = (lane >> 4) << 3;               // 0 or 8
  const _Float16* p = base + col * ld + kb;
  v8h lo = *reinterpret_cast<const v8h*>(p);
  v8h hi = *reinterpret_cast<const v8h*>(p + 16);
  v16h f;
#pragma unroll
  for (int i = 0; i < 8; ++i) { f[i] = lo[i]; f[i + 8] = hi[i]; }
  return f;
}

__global__ void cvt_f32_f16(const v4f* __restrict__ in, v4h* __restrict__ out, int n4) {
  int i = blockIdx.x * blockDim.x + threadIdx.x;
  if (i < n4) {
    v4f x = in[i];
    v4h y;
#pragma unroll
    for (int j = 0; j < 4; ++j) y[j] = (_Float16)x[j];
    out[i] = y;
  }
}

// ---------------------------------------------------------------------------
// Shared GEMM tile stage: A 128x32 halves (2 chunks/thread), B 256x32 halves
// (4 chunks/thread) => 6 async ops per thread per stage.
// ---------------------------------------------------------------------------
#define GEMM_LDA 40
#define GEMM_LDB 40
#define GEMM_STAGE_OPS 6

static __device__ __forceinline__ void stage_gemm_tiles(
    _Float16* As, _Float16* Bs,
    const _Float16* __restrict__ X, const _Float16* __restrict__ W,
    int mb, int nb, int k0, int tid) {
#pragma unroll
  for (int i = 0; i < 2; ++i) {             // A: 128 rows x 32 halves
    int c = tid + i * 256;
    int r = c >> 2, off = (c & 3) * 8;
    stage16(&As[r * GEMM_LDA + off], &X[(size_t)(mb + r) * EMB + k0 + off]);
  }
#pragma unroll
  for (int i = 0; i < 4; ++i) {             // B: 256 rows x 32 halves
    int c = tid + i * 256;
    int r = c >> 2, off = (c & 3) * 8;
    stage16(&Bs[r * GEMM_LDB + off], &W[(size_t)(nb + r) * EMB + k0 + off]);
  }
}

// ---------------------------------------------------------------------------
// QKV projection: qkv[m][n] = sum_k X[m][k] * Wqkv[n][k] + b[n]
// Block tile 128(M) x 256(N), 8 waves in 2x4 grid, each wave 64x64 (4x4 WMMA).
// Double-buffered LDS; results scattered f16 into Q[h][s][d], K[h][s][d],
// V^T[h][d][s]. Each 16-wide N-tile stays inside one segment & head.
// ---------------------------------------------------------------------------
__global__ __launch_bounds__(256) void qkv_gemm(
    const _Float16* __restrict__ X, const _Float16* __restrict__ W,
    const float* __restrict__ bias,
    _Float16* __restrict__ Qb, _Float16* __restrict__ Kb, _Float16* __restrict__ VTb) {
  __shared__ __align__(16) _Float16 As[2][128 * GEMM_LDA];
  __shared__ __align__(16) _Float16 Bs[2][256 * GEMM_LDB];

  int tid = threadIdx.x, lane = tid & 31, wid = tid >> 5;
  int wm = (wid & 1) * 64, wn = (wid >> 1) * 64;
  int mb = blockIdx.y * 128, nb = blockIdx.x * 256;

  v8f acc[4][4];
#pragma unroll
  for (int i = 0; i < 4; ++i)
#pragma unroll
    for (int j = 0; j < 4; ++j) acc[i][j] = {};

  const int NIT = EMB / 32;
  stage_gemm_tiles(As[0], Bs[0], X, W, mb, nb, 0, tid);
  for (int it = 0; it < NIT; ++it) {
    int cur = it & 1;
    bool has_next = (it + 1) < NIT;
    if (has_next)
      stage_gemm_tiles(As[cur ^ 1], Bs[cur ^ 1], X, W, mb, nb, (it + 1) * 32, tid);
    if (has_next) { WAIT_ASYNC(GEMM_STAGE_OPS); } else { WAIT_ASYNC(0); }
    __syncthreads();

    v16h a[4], b[4];
#pragma unroll
    for (int t = 0; t < 4; ++t) a[t] = frag_a_16(&As[cur][(wm + t * 16) * GEMM_LDA], GEMM_LDA, lane);
#pragma unroll
    for (int t = 0; t < 4; ++t) b[t] = frag_b_16(&Bs[cur][(wn + t * 16) * GEMM_LDB], GEMM_LDB, lane);
#pragma unroll
    for (int i = 0; i < 4; ++i)
#pragma unroll
      for (int j = 0; j < 4; ++j) acc[i][j] = wmma_f16(a[i], b[j], acc[i][j]);
    __syncthreads();
  }

  int hlf = lane >> 4, nl = lane & 15;
#pragma unroll
  for (int j = 0; j < 4; ++j) {
    int n0 = nb + wn + j * 16;              // 16-aligned: one segment, one head
    float bj = bias[n0 + nl];
    int seg = n0 / EMB;
    int col0 = n0 - seg * EMB;
    int h = col0 >> 6;
    int d = (col0 & 63) + nl;
    _Float16* dst;
    size_t rstride;
    if (seg == 0)      { dst = Qb  + (size_t)h * S_LEN * DHD + d;       rstride = DHD; }
    else if (seg == 1) { dst = Kb  + (size_t)h * S_LEN * DHD + d;       rstride = DHD; }
    else               { dst = VTb + ((size_t)h * DHD + d) * S_LEN;     rstride = 1;   }
#pragma unroll
    for (int i = 0; i < 4; ++i)
#pragma unroll
      for (int r = 0; r < 8; ++r) {
        int m = mb + wm + i * 16 + r + hlf * 8;
        dst[(size_t)m * rstride] = (_Float16)(acc[i][j][r] + bj);
      }
  }
}

// ---------------------------------------------------------------------------
// Flash attention (causal, no 1/sqrt(dh) scale per reference).
// Block: 128 threads (4 waves), 64 query rows; 16 rows per wave.
// 64-wide key tiles, double-buffered async K/V staging, fp32 online softmax.
// ---------------------------------------------------------------------------
#define FLASH_LD 72                          // 64 + 8 halves pad (144B rows)
#define FLASH_STAGE_OPS 8

static __device__ __forceinline__ void stage_kv_tiles(
    _Float16* Kt, _Float16* Vt,
    const _Float16* __restrict__ Kb, const _Float16* __restrict__ VTb,
    int h, int kb0, int tid) {
#pragma unroll
  for (int i = 0; i < 4; ++i) {             // 512 chunks of 8 halves each
    int c = tid + i * 128;
    int r = c >> 3, off = (c & 7) * 8;
    stage16(&Kt[r * FLASH_LD + off], &Kb[((size_t)h * S_LEN + kb0 + r) * DHD + off]);
    stage16(&Vt[r * FLASH_LD + off], &VTb[((size_t)h * DHD + r) * S_LEN + kb0 + off]);
  }
}

__global__ __launch_bounds__(128) void flash_attn(
    const _Float16* __restrict__ Qb, const _Float16* __restrict__ Kb,
    const _Float16* __restrict__ VTb, _Float16* __restrict__ Ab) {
  __shared__ __align__(16) _Float16 Kt[2][64 * FLASH_LD];
  __shared__ __align__(16) _Float16 Vt[2][64 * FLASH_LD];
  __shared__ __align__(16) _Float16 Pt[4 * 16 * FLASH_LD];

  int tid = threadIdx.x, lane = tid & 31, wid = tid >> 5;
  int h = blockIdx.y;
  int qbase = blockIdx.x * 64;
  int qrow0 = qbase + wid * 16;
  int hlf = lane >> 4, nl = lane & 15;

  v16h qf[2];                               // Q 16x64 in A-fragments
  {
    const _Float16* qp = Qb + ((size_t)h * S_LEN + qrow0) * DHD;
    qf[0] = frag_a_16(qp, DHD, lane);
    qf[1] = frag_a_16(qp + 32, DHD, lane);
  }

  v8f o[4];
#pragma unroll
  for (int t = 0; t < 4; ++t) o[t] = {};
  float mrow[8], lrow[8];
#pragma unroll
  for (int r = 0; r < 8; ++r) { mrow[r] = -__builtin_inff(); lrow[r] = 0.f; }

  const int nkb = (qbase >> 6) + 1;         // causal: key tiles 0 .. qbase/64
  stage_kv_tiles(Kt[0], Vt[0], Kb, VTb, h, 0, tid);
  for (int it = 0; it < nkb; ++it) {
    int cur = it & 1;
    int kb0 = it * 64;
    bool has_next = (it + 1) < nkb;
    if (has_next)
      stage_kv_tiles(Kt[cur ^ 1], Vt[cur ^ 1], Kb, VTb, h, kb0 + 64, tid);
    if (has_next) { WAIT_ASYNC(FLASH_STAGE_OPS); } else { WAIT_ASYNC(0); }
    __syncthreads();

    v8f s[4];                               // scores 16x64 (4 N-tiles)
#pragma unroll
    for (int nt = 0; nt < 4; ++nt) {
      v8f c = {};
      c = wmma_f16(qf[0], frag_b_16(&Kt[cur][nt * 16 * FLASH_LD], FLASH_LD, lane), c);
      c = wmma_f16(qf[1], frag_b_16(&Kt[cur][nt * 16 * FLASH_LD] + 32, FLASH_LD, lane), c);
      s[nt] = c;
    }

    float rmax[8];
#pragma unroll
    for (int r = 0; r < 8; ++r) rmax[r] = mrow[r];
#pragma unroll
    for (int nt = 0; nt < 4; ++nt) {
      int kcol = kb0 + nt * 16 + nl;
#pragma unroll
      for (int r = 0; r < 8; ++r) {
        int q = qrow0 + r + hlf * 8;
        if (kcol > q) s[nt][r] = -__builtin_inff();   // causal mask
        rmax[r] = fmaxf(rmax[r], s[nt][r]);
      }
    }
#pragma unroll
    for (int off = 1; off < 16; off <<= 1)
#pragma unroll
      for (int r = 0; r < 8; ++r)
        rmax[r] = fmaxf(rmax[r], __shfl_xor(rmax[r], off, 32));

    float scale[8], psum[8];
#pragma unroll
    for (int r = 0; r < 8; ++r) {
      scale[r] = __expf(mrow[r] - rmax[r]); // first iter: exp(-inf)=0
      mrow[r] = rmax[r];
      psum[r] = 0.f;
    }

    _Float16* Pw = &Pt[wid * 16 * FLASH_LD];
#pragma unroll
    for (int nt = 0; nt < 4; ++nt)
#pragma unroll
      for (int r = 0; r < 8; ++r) {
        float p = __expf(s[nt][r] - mrow[r]);
        psum[r] += p;
        Pw[(r + hlf * 8) * FLASH_LD + nt * 16 + nl] = (_Float16)p;
      }
#pragma unroll
    for (int off = 1; off < 16; off <<= 1)
#pragma unroll
      for (int r = 0; r < 8; ++r) psum[r] += __shfl_xor(psum[r], off, 32);
#pragma unroll
    for (int r = 0; r < 8; ++r) lrow[r] = lrow[r] * scale[r] + psum[r];

#pragma unroll
    for (int nt = 0; nt < 4; ++nt)
#pragma unroll
      for (int r = 0; r < 8; ++r) o[nt][r] *= scale[r];

    v16h pf0 = frag_a_16(Pw, FLASH_LD, lane);   // P 16x64 re-packed via LDS
    v16h pf1 = frag_a_16(Pw + 32, FLASH_LD, lane);
#pragma unroll
    for (int nt = 0; nt < 4; ++nt) {
      o[nt] = wmma_f16(pf0, frag_b_16(&Vt[cur][nt * 16 * FLASH_LD], FLASH_LD, lane), o[nt]);
      o[nt] = wmma_f16(pf1, frag_b_16(&Vt[cur][nt * 16 * FLASH_LD] + 32, FLASH_LD, lane), o[nt]);
    }
    __syncthreads();
  }

#pragma unroll
  for (int r = 0; r < 8; ++r) lrow[r] = 1.0f / lrow[r];
#pragma unroll
  for (int nt = 0; nt < 4; ++nt)
#pragma unroll
    for (int r = 0; r < 8; ++r) {
      int m = qrow0 + r + hlf * 8;
      int d = nt * 16 + nl;
      Ab[(size_t)m * EMB + h * DHD + d] = (_Float16)(o[nt][r] * lrow[r]);
    }
}

// ---------------------------------------------------------------------------
// Output projection: out[m][n] = sum_k A[m][k] * Wout[n][k] + b[n]  (fp32 out)
// ---------------------------------------------------------------------------
__global__ __launch_bounds__(256) void out_gemm(
    const _Float16* __restrict__ A, const _Float16* __restrict__ W,
    const float* __restrict__ bias, float* __restrict__ out) {
  __shared__ __align__(16) _Float16 As[2][128 * GEMM_LDA];
  __shared__ __align__(16) _Float16 Bs[2][256 * GEMM_LDB];

  int tid = threadIdx.x, lane = tid & 31, wid = tid >> 5;
  int wm = (wid & 1) * 64, wn = (wid >> 1) * 64;
  int mb = blockIdx.y * 128, nb = blockIdx.x * 256;

  v8f acc[4][4];
#pragma unroll
  for (int i = 0; i < 4; ++i)
#pragma unroll
    for (int j = 0; j < 4; ++j) acc[i][j] = {};

  const int NIT = EMB / 32;
  stage_gemm_tiles(As[0], Bs[0], A, W, mb, nb, 0, tid);
  for (int it = 0; it < NIT; ++it) {
    int cur = it & 1;
    bool has_next = (it + 1) < NIT;
    if (has_next)
      stage_gemm_tiles(As[cur ^ 1], Bs[cur ^ 1], A, W, mb, nb, (it + 1) * 32, tid);
    if (has_next) { WAIT_ASYNC(GEMM_STAGE_OPS); } else { WAIT_ASYNC(0); }
    __syncthreads();

    v16h a[4], b[4];
#pragma unroll
    for (int t = 0; t < 4; ++t) a[t] = frag_a_16(&As[cur][(wm + t * 16) * GEMM_LDA], GEMM_LDA, lane);
#pragma unroll
    for (int t = 0; t < 4; ++t) b[t] = frag_b_16(&Bs[cur][(wn + t * 16) * GEMM_LDB], GEMM_LDB, lane);
#pragma unroll
    for (int i = 0; i < 4; ++i)
#pragma unroll
      for (int j = 0; j < 4; ++j) acc[i][j] = wmma_f16(a[i], b[j], acc[i][j]);
    __syncthreads();
  }

  int hlf = lane >> 4, nl = lane & 15;
  float bj[4];
#pragma unroll
  for (int j = 0; j < 4; ++j) bj[j] = bias[nb + wn + j * 16 + nl];
#pragma unroll
  for (int i = 0; i < 4; ++i)
#pragma unroll
    for (int r = 0; r < 8; ++r) {
      int m = mb + wm + i * 16 + r + hlf * 8;
      float* row = out + (size_t)m * EMB + nb + wn + nl;
#pragma unroll
      for (int j = 0; j < 4; ++j) row[j * 16] = acc[i][j][r] + bj[j];
    }
}

extern "C" void kernel_launch(void* const* d_in, const int* in_sizes, int n_in,
                              void* d_out, int out_size, void* d_ws, size_t ws_size,
                              hipStream_t stream) {
  const float* x     = (const float*)d_in[0];
  // d_in[1] = causal mask (bool) — handled analytically (k <= q)
  const float* w_qkv = (const float*)d_in[2];
  const float* b_qkv = (const float*)d_in[3];
  const float* w_out = (const float*)d_in[4];
  const float* b_out = (const float*)d_in[5];
  float* out = (float*)d_out;

  // f16 workspace layout (~36.2 MB total)
  _Float16* ws = (_Float16*)d_ws;
  size_t off = 0;
  _Float16* Xh    = ws + off; off += (size_t)S_LEN * EMB;
  _Float16* Wqkvh = ws + off; off += (size_t)E3 * EMB;
  _Float16* Wouth = ws + off; off += (size_t)EMB * EMB;
  _Float16* Qh    = ws + off; off += (size_t)NH * S_LEN * DHD;
  _Float16* Kh    = ws + off; off += (size_t)NH * S_LEN * DHD;
  _Float16* VTh   = ws + off; off += (size_t)NH * S_LEN * DHD;
  _Float16* Ah    = ws + off; off += (size_t)S_LEN * EMB;

  int nx4 = S_LEN * EMB / 4, nwq4 = E3 * EMB / 4, nwo4 = EMB * EMB / 4;
  cvt_f32_f16<<<(nx4  + 255) / 256, 256, 0, stream>>>((const v4f*)x,     (v4h*)Xh,    nx4);
  cvt_f32_f16<<<(nwq4 + 255) / 256, 256, 0, stream>>>((const v4f*)w_qkv, (v4h*)Wqkvh, nwq4);
  cvt_f32_f16<<<(nwo4 + 255) / 256, 256, 0, stream>>>((const v4f*)w_out, (v4h*)Wouth, nwo4);

  qkv_gemm  <<<dim3(E3 / 256, S_LEN / 128), 256, 0, stream>>>(Xh, Wqkvh, b_qkv, Qh, Kh, VTh);
  flash_attn<<<dim3(S_LEN / 64, NH),        128, 0, stream>>>(Qh, Kh, VTh, Ah);
  out_gemm  <<<dim3(EMB / 256, S_LEN / 128), 256, 0, stream>>>(Ah, Wouth, b_out, out);
}
